// ObjectRetriever_20177756357024
// MI455X (gfx1250) — matrix operand order repaired and checked by reading, not compiled, not run on hardware
//
#include <hip/hip_runtime.h>
#include <hip/hip_bf16.h>

// ---------------------------------------------------------------------------
// ObjectRetriever pipeline for MI455X (gfx1250, wave32, WMMA).
// All heavy GEMMs: f16 operands (pre-converted), f32 accumulate via
// v_wmma_f32_16x16x32_f16. Single GEMM path (B always [N][K] row-major, i.e.
// A @ W^T); V matrices are pre-transposed so attn@V also uses this path.
// ---------------------------------------------------------------------------

typedef __attribute__((ext_vector_type(16))) _Float16 v16h;
typedef __attribute__((ext_vector_type(8)))  _Float16 v8h;
typedef __attribute__((ext_vector_type(4)))  _Float16 v4h;
typedef __attribute__((ext_vector_type(8)))  float    v8f;

#define C_NHEAD    4
#define C_NOBJ     4096
#define C_NFRAMES  64
#define C_LWIN     192
#define C_NW       62
#define C_EDIM     1024
#define C_IMGDIM   2048
#define C_DFF      2048
#define C_HD       256               // head dim = 1024/4
#define C_ROWS     (C_NW * C_LWIN)   // 11904

// ---------------------------------------------------------------------------
// Batched WMMA GEMM, f16 in / f32 accumulate:
//   val = alpha * sum_k A[m][k]*B[n][k] + bias[n] + resid[m][n]; optional ReLU
//   C (f32) and Ch (f16) outputs are each optional.
// Batching: z = blockIdx.z = w*nh + h; element offsets w*{a,b,c}W + h*{a,b,c}H
// (cW/cH apply to both C and Ch). Block = 128 thr = 4 waves; wave = 16Mx64N.
// Requires M%(64)==0 per grid.y, N%64==0, K%32==0; 32B-aligned B rows,
// 16B-aligned A rows (true for all call sites).
// ---------------------------------------------------------------------------
__global__ __launch_bounds__(128)
void gemm_wmma_h(const _Float16* __restrict__ A, const _Float16* __restrict__ B,
                 const float* __restrict__ bias, const float* __restrict__ resid,
                 float* __restrict__ C, _Float16* __restrict__ Ch,
                 int M, int N, int K, int lda, int ldb, int ldc,
                 long aW, long aH, long bW, long bH, long cW, long cH,
                 int nh, float alpha, int act)
{
    const int z = blockIdx.z;
    const int w = z / nh, h = z % nh;
    A += (long)w * aW + (long)h * aH;
    B += (long)w * bW + (long)h * bH;
    const long coff = (long)w * cW + (long)h * cH;
    if (C)  C  += coff;
    if (Ch) Ch += coff;

    const int wave = threadIdx.x >> 5;       // 0..3 along M
    const int lane = threadIdx.x & 31;
    const int l    = lane & 15;
    const int hi   = lane >> 4;

    const int m0 = blockIdx.y * 64 + wave * 16;
    const int n0 = blockIdx.x * 64;

    v8f acc0 = {}, acc1 = {}, acc2 = {}, acc3 = {};

    // 32-bit index arithmetic (all buffers < 2^31 elements).
    const _Float16* Arow = A + (m0 + l) * lda;
    const _Float16* Bbase = B + (n0 + l) * ldb;   // + t*16*ldb per N tile

    for (int k0 = 0; k0 < K; k0 += 32) {
        // A fragment 16x32: lane l holds row m0+l, K = {hi*8+0..7, 16+hi*8+0..7}
        const int kb = k0 + hi * 8;
        const v8h a0 = *(const v8h*)(Arow + kb);
        const v8h a1 = *(const v8h*)(Arow + kb + 16);
        v16h af;
        #pragma unroll
        for (int j = 0; j < 8; ++j) { af[j] = a0[j]; af[8 + j] = a1[j]; }

        const int kbb = k0 + hi * 8;             // B: K = {hi*8+0..7, 16+hi*8+0..7}
        #pragma unroll
        for (int t = 0; t < 4; ++t) {
            const v8h b0 = *(const v8h*)(Bbase + t * 16 * ldb + kbb);
            const v8h b1 = *(const v8h*)(Bbase + t * 16 * ldb + kbb + 16);
            v16h bf;
            #pragma unroll
            for (int j = 0; j < 8; ++j) { bf[j] = b0[j]; bf[8 + j] = b1[j]; }
            v8f* accp = (t == 0) ? &acc0 : (t == 1) ? &acc1 : (t == 2) ? &acc2 : &acc3;
            *accp = __builtin_amdgcn_wmma_f32_16x16x32_f16(
                        false, af, false, bf, (short)0, *accp, false, false);
        }
    }

    // Epilogue: C/D 16x16 f32 layout: lane l -> col, VGPR r -> row hi*8+r.
    #pragma unroll
    for (int t = 0; t < 4; ++t) {
        const v8f acc = (t == 0) ? acc0 : (t == 1) ? acc1 : (t == 2) ? acc2 : acc3;
        const int n = n0 + t * 16 + l;
        const float bv = bias ? bias[n] : 0.0f;
        #pragma unroll
        for (int r = 0; r < 8; ++r) {
            const int m = m0 + hi * 8 + r;
            float v = acc[r] * alpha + bv;
            if (resid) v += resid[m * ldc + n];
            if (act)   v = v > 0.0f ? v : 0.0f;
            if (C)  C[m * ldc + n]  = v;
            if (Ch) Ch[m * ldc + n] = (_Float16)v;
        }
    }
}

// ---------------------------------------------------------------------------
// Row softmax (fp32 in, f16 out) with optional key-padding mask.
// One block (256 threads) per row; masked tail -> exact zeros.
// ---------------------------------------------------------------------------
__global__ __launch_bounds__(256)
void softmax_rows(const float* __restrict__ S, _Float16* __restrict__ P, int cols,
                  const int* __restrict__ lenPer, int rowsPerW)
{
    const int row = blockIdx.x;
    const float* p = S + (long)row * cols;
    _Float16* q = P + (long)row * cols;
    int len = lenPer ? lenPer[row / rowsPerW] : cols;
    if (len < 1) len = 1;
    if (len > cols) len = cols;

    __shared__ float red[256];
    const int tid = threadIdx.x;

    float mx = -3.4e38f;
    for (int j = tid; j < len; j += 256) mx = fmaxf(mx, p[j]);
    red[tid] = mx; __syncthreads();
    for (int s = 128; s > 0; s >>= 1) {
        if (tid < s) red[tid] = fmaxf(red[tid], red[tid + s]);
        __syncthreads();
    }
    mx = red[0]; __syncthreads();

    float sum = 0.0f;
    for (int j = tid; j < len; j += 256) sum += __expf(p[j] - mx);
    red[tid] = sum; __syncthreads();
    for (int s = 128; s > 0; s >>= 1) {
        if (tid < s) red[tid] += red[tid + s];
        __syncthreads();
    }
    const float inv = 1.0f / red[0];
    for (int j = tid; j < cols; j += 256)
        q[j] = (j < len) ? (_Float16)(__expf(p[j] - mx) * inv) : (_Float16)0.0f;
}

// ---------------------------------------------------------------------------
// In-place LayerNorm (fp32) + optional f16 copy. dim % 256 == 0, dim <= 2048.
// ---------------------------------------------------------------------------
__global__ __launch_bounds__(256)
void layernorm_rows(float* __restrict__ X, _Float16* __restrict__ Xh,
                    const float* __restrict__ g, const float* __restrict__ beta, int dim)
{
    const int row = blockIdx.x;
    float* x = X + (long)row * dim;
    __shared__ float red[256];
    const int tid = threadIdx.x;
    const int per = dim / 256;

    float lv[8];
    float s = 0.0f;
    for (int t = 0; t < per; ++t) { lv[t] = x[tid + t * 256]; s += lv[t]; }
    red[tid] = s; __syncthreads();
    for (int k = 128; k > 0; k >>= 1) {
        if (tid < k) red[tid] += red[tid + k];
        __syncthreads();
    }
    const float mean = red[0] / (float)dim; __syncthreads();

    float vs = 0.0f;
    for (int t = 0; t < per; ++t) { const float d = lv[t] - mean; vs += d * d; }
    red[tid] = vs; __syncthreads();
    for (int k = 128; k > 0; k >>= 1) {
        if (tid < k) red[tid] += red[tid + k];
        __syncthreads();
    }
    const float inv = rsqrtf(red[0] / (float)dim + 1e-5f);
    for (int t = 0; t < per; ++t) {
        const int j = tid + t * 256;
        const float v = (lv[t] - mean) * inv * g[j] + beta[j];
        x[j] = v;
        if (Xh) Xh[(long)row * dim + j] = (_Float16)v;
    }
}

// f32 -> f16 bulk convert; n % 1024 == 0.
__global__ __launch_bounds__(256)
void cvt_f32_f16(const float* __restrict__ in, _Float16* __restrict__ out, int n)
{
    const int i = (blockIdx.x * 256 + threadIdx.x) * 4;
    if (i < n) {
        const float4 v = *(const float4*)(in + i);
        v4h o; o[0] = (_Float16)v.x; o[1] = (_Float16)v.y;
        o[2] = (_Float16)v.z; o[3] = (_Float16)v.w;
        *(v4h*)(out + i) = o;
    }
}

// V transpose: out[z][d][k] = in[(w*rowsPerW + k)*inLd + colBase + h*HD + d]
// out block per z is HD x K, row-major.  z = w*nh + h.
__global__ __launch_bounds__(256)
void transpose_v(const _Float16* __restrict__ in, _Float16* __restrict__ out,
                 int inLd, int rowsPerW, int colBase, int K, int nh)
{
    const int z = blockIdx.y;
    const int w = z / nh, h = z % nh;
    const _Float16* src = in + (long)(w * rowsPerW) * inLd + colBase + h * C_HD;
    _Float16* dst = out + (long)z * C_HD * K;
    const int e = blockIdx.x * 256 + threadIdx.x;   // over HD*K
    const int d = e / K, k = e % K;
    dst[d * K + k] = src[(long)k * inLd + d];
}

// Histogram obj_idx -> cum[0..nframes], lenw[w] = cum[w+3]-cum[w].
__global__ __launch_bounds__(256)
void frame_cumsum(const int* __restrict__ obj_idx, int n,
                  int* __restrict__ cum, int* __restrict__ lenw)
{
    __shared__ int cnt[C_NFRAMES];
    if (threadIdx.x < C_NFRAMES) cnt[threadIdx.x] = 0;
    __syncthreads();
    for (int i = threadIdx.x; i < n; i += 256) {
        int f = obj_idx[i];
        if (f >= 0 && f < C_NFRAMES) atomicAdd(&cnt[f], 1);
    }
    __syncthreads();
    if (threadIdx.x == 0) {
        int acc = 0; cum[0] = 0;
        for (int f = 0; f < C_NFRAMES; ++f) { acc += cnt[f]; cum[f + 1] = acc; }
        for (int w = 0; w < C_NW; ++w) lenw[w] = cum[w + 3] - cum[w];
    }
}

// obj_mod[i] = 0.5*obj[i] + 0.5*cross64[obj_idx[i]]
__global__ __launch_bounds__(256)
void obj_mod_kernel(const float* __restrict__ obj, const float* __restrict__ cross64,
                    const int* __restrict__ idx, float* __restrict__ out)
{
    const int i = blockIdx.x;
    const int f = idx[i];
    const float* a = obj + (long)i * C_EDIM;
    const float* c = cross64 + (long)f * C_EDIM;
    float* o = out + (long)i * C_EDIM;
    for (int j = threadIdx.x; j < C_EDIM; j += 256) o[j] = 0.5f * a[j] + 0.5f * c[j];
}

// X[w*L_WIN + slot][:] = valid ? obj_mod[g] + pos_emb[pid] : 0   (f32 + f16)
__global__ __launch_bounds__(256)
void build_windows(const float* __restrict__ objmod, const float* __restrict__ pos,
                   const int* __restrict__ cum, float* __restrict__ X,
                   _Float16* __restrict__ Xh)
{
    const int r = blockIdx.x;                 // 0 .. NW*L_WIN-1
    const int wnd = r / C_LWIN, slot = r % C_LWIN;
    const int start = cum[wnd];
    const int len = cum[wnd + 3] - start;
    const int t1 = cum[wnd + 1] - start;
    const int t2 = cum[wnd + 2] - start;
    const bool valid = slot < len;
    int g = start + slot;
    g = g < 0 ? 0 : (g > C_NOBJ - 1 ? C_NOBJ - 1 : g);
    const int pid = slot < t1 ? 0 : (slot < t2 ? 1 : 2);
    float* x = X + (long)r * C_EDIM;
    _Float16* xh = Xh + (long)r * C_EDIM;
    const float* s  = objmod + (long)g * C_EDIM;
    const float* pe = pos + (long)pid * C_EDIM;
    for (int j = threadIdx.x; j < C_EDIM; j += 256) {
        const float v = valid ? (s[j] + pe[j]) : 0.0f;
        x[j] = v; xh[j] = (_Float16)v;
    }
}

// out[i] = X[clip(obj_idx[i]-1,0,NW-1)*L_WIN + (i - cum[win])]
__global__ __launch_bounds__(256)
void gather_out(const float* __restrict__ X, const int* __restrict__ idx,
                const int* __restrict__ cum, float* __restrict__ out)
{
    const int i = blockIdx.x;
    int win = idx[i] - 1;
    win = win < 0 ? 0 : (win > C_NW - 1 ? C_NW - 1 : win);
    const int slot = i - cum[win];
    const float* s = X + ((long)win * C_LWIN + slot) * C_EDIM;
    float* o = out + (long)i * C_EDIM;
    for (int j = threadIdx.x; j < C_EDIM; j += 256) o[j] = s[j];
}

// ---------------------------------------------------------------------------
// Host side
// ---------------------------------------------------------------------------
static void launch_gemm(hipStream_t s,
                        const _Float16* A, const _Float16* B, const float* bias,
                        const float* resid, float* C, _Float16* Ch,
                        int M, int N, int K, int lda, int ldb, int ldc,
                        float alpha, int act,
                        int Z = 1, int nh = 1,
                        long aW = 0, long aH = 0, long bW = 0, long bH = 0,
                        long cW = 0, long cH = 0)
{
    dim3 grid((unsigned)(N / 64), (unsigned)(M / 64), (unsigned)Z);
    hipLaunchKernelGGL(gemm_wmma_h, grid, dim3(128), 0, s,
                       A, B, bias, resid, C, Ch, M, N, K, lda, ldb, ldc,
                       aW, aH, bW, bH, cW, cH, nh, alpha, act);
}

static void cvt(hipStream_t s, const float* in, _Float16* out, size_t n)
{
    hipLaunchKernelGGL(cvt_f32_f16, dim3((unsigned)(n / 1024)), dim3(256), 0, s,
                       in, out, (int)n);
}

extern "C" void kernel_launch(void* const* d_in, const int* in_sizes, int n_in,
                              void* d_out, int out_size, void* d_ws, size_t ws_size,
                              hipStream_t stream)
{
    (void)in_sizes; (void)n_in; (void)out_size;
    const float* img      = (const float*)d_in[0];
    const float* obj      = (const float*)d_in[1];
    const int*   obj_idx  = (const int*)  d_in[2];
    const float* w_i2o    = (const float*)d_in[3];
    const float* b_i2o    = (const float*)d_in[4];
    const float* cin_w    = (const float*)d_in[5];
    const float* cin_b    = (const float*)d_in[6];
    const float* cout_w   = (const float*)d_in[7];
    const float* cout_b   = (const float*)d_in[8];
    const float* sa_in_w  = (const float*)d_in[9];
    const float* sa_in_b  = (const float*)d_in[10];
    const float* sa_out_w = (const float*)d_in[11];
    const float* sa_out_b = (const float*)d_in[12];
    const float* sa_l1_w  = (const float*)d_in[13];
    const float* sa_l1_b  = (const float*)d_in[14];
    const float* sa_l2_w  = (const float*)d_in[15];
    const float* sa_l2_b  = (const float*)d_in[16];
    const float* sa_n1_g  = (const float*)d_in[17];
    const float* sa_n1_b  = (const float*)d_in[18];
    const float* sa_n2_g  = (const float*)d_in[19];
    const float* sa_n2_b  = (const float*)d_in[20];
    const float* pos_emb  = (const float*)d_in[21];
    float* out = (float*)d_out;

    // ---- workspace bump allocator (256B aligned) ----
    char* base = (char*)d_ws;
    size_t off = 0;
    auto alloc = [&](size_t bytes) -> char* {
        char* p = base + off;
        off += (bytes + 255) & ~(size_t)255;
        return p;
    };
    const size_t E = C_EDIM;
    int*      iCum = (int*)alloc(65 * sizeof(int));
    int*      iLen = (int*)alloc(C_NW * sizeof(int));
    // f16 weights
    _Float16* Wi2o = (_Float16*)alloc((size_t)E * C_IMGDIM * 2);
    _Float16* Wcin = (_Float16*)alloc((size_t)3 * E * E * 2);
    _Float16* Wcout= (_Float16*)alloc((size_t)E * E * 2);
    _Float16* Wsin = (_Float16*)alloc((size_t)2 * 3 * E * E * 2);
    _Float16* Wsout= (_Float16*)alloc((size_t)2 * E * E * 2);
    _Float16* Wl1  = (_Float16*)alloc((size_t)2 * C_DFF * E * 2);
    _Float16* Wl2  = (_Float16*)alloc((size_t)2 * E * C_DFF * 2);
    // f16 inputs
    _Float16* imgH = (_Float16*)alloc((size_t)64 * C_IMGDIM * 2);
    _Float16* objH = (_Float16*)alloc((size_t)C_NOBJ * E * 2);
    // small cross temporaries
    _Float16* P64h = (_Float16*)alloc((size_t)64 * E * 2);
    _Float16* Q64h = (_Float16*)alloc((size_t)64 * E * 2);
    _Float16* O64h = (_Float16*)alloc((size_t)64 * E * 2);
    float*    X64  = (float*)  alloc((size_t)64 * E * 4);
    // big buffers
    _Float16* QKVh = (_Float16*)alloc((size_t)C_ROWS * 3 * E * 2);          // 73 MB
    float*    Sbuf = (float*)  alloc((size_t)C_NW * C_NHEAD * C_LWIN * C_LWIN * 4); // 36.6 MB
    _Float16* Pw   = (_Float16*)alloc((size_t)C_NW * C_NHEAD * C_LWIN * C_LWIN * 2);
    _Float16* Vt   = (_Float16*)alloc((size_t)C_NW * C_NHEAD * C_HD * C_LWIN * 2);
    _Float16* Obh  = (_Float16*)alloc((size_t)C_ROWS * E * 2);
    float*    Tbuf = (float*)  alloc((size_t)C_ROWS * E * 4);
    _Float16* Th   = (_Float16*)alloc((size_t)C_ROWS * E * 2);
    float*    Xbuf = (float*)  alloc((size_t)C_ROWS * E * 4);
    _Float16* Xh   = (_Float16*)alloc((size_t)C_ROWS * E * 2);
    float*    Ubuf = (float*)  alloc((size_t)C_ROWS * E * 4);
    _Float16* Uh   = (_Float16*)alloc((size_t)C_ROWS * E * 2);
    _Float16* H1h  = (_Float16*)alloc((size_t)C_ROWS * C_DFF * 2);          // 48.8 MB
    if (off > ws_size) return;  // workspace too small; bail deterministically

    // Cross-stage temporaries overlap later-used big buffers (dead before reuse).
    _Float16* Kch    = QKVh;                              // (4096,1024) f16
    _Float16* Vch    = QKVh + (size_t)C_NOBJ * E;         // (4096,1024) f16
    _Float16* Vtc    = QKVh + (size_t)2 * C_NOBJ * E;     // (4,256,4096) f16
    float*    Scross = Sbuf;                              // (4,64,4096) f32
    _Float16* Pch    = Pw;                                // (4,64,4096) f16
    float*    OMod   = Tbuf;                              // (4096,1024) f32

    const float sc = 1.0f / 16.0f;   // 1/sqrt(head_dim=256)

    // 0) frame histogram / cumsum; convert weights + inputs to f16
    hipLaunchKernelGGL(frame_cumsum, dim3(1), dim3(256), 0, stream,
                       obj_idx, C_NOBJ, iCum, iLen);
    cvt(stream, w_i2o,    Wi2o,  (size_t)E * C_IMGDIM);
    cvt(stream, cin_w,    Wcin,  (size_t)3 * E * E);
    cvt(stream, cout_w,   Wcout, (size_t)E * E);
    cvt(stream, sa_in_w,  Wsin,  (size_t)2 * 3 * E * E);
    cvt(stream, sa_out_w, Wsout, (size_t)2 * E * E);
    cvt(stream, sa_l1_w,  Wl1,   (size_t)2 * C_DFF * E);
    cvt(stream, sa_l2_w,  Wl2,   (size_t)2 * E * C_DFF);
    cvt(stream, img,      imgH,  (size_t)64 * C_IMGDIM);
    cvt(stream, obj,      objH,  (size_t)C_NOBJ * E);

    // 1) per-frame image projection: P64 = img @ w_i2o^T + b      (f16 out)
    launch_gemm(stream, imgH, Wi2o, b_i2o, nullptr, nullptr, P64h,
                64, C_EDIM, C_IMGDIM, C_IMGDIM, C_IMGDIM, C_EDIM, 1.0f, 0);

    // 2) cross Q/K/V projections (f16 out)
    launch_gemm(stream, P64h, Wcin, cin_b, nullptr, nullptr, Q64h,
                64, C_EDIM, C_EDIM, C_EDIM, C_EDIM, C_EDIM, 1.0f, 0);
    launch_gemm(stream, objH, Wcin + (size_t)E * E, cin_b + E, nullptr, nullptr, Kch,
                C_NOBJ, C_EDIM, C_EDIM, C_EDIM, C_EDIM, C_EDIM, 1.0f, 0);
    launch_gemm(stream, objH, Wcin + (size_t)2 * E * E, cin_b + 2 * E, nullptr, nullptr, Vch,
                C_NOBJ, C_EDIM, C_EDIM, C_EDIM, C_EDIM, C_EDIM, 1.0f, 0);

    // 3) cross scores (4 heads): S_h = (Q_h @ K_h^T)/16 -> f32; softmax -> f16
    launch_gemm(stream, Q64h, Kch, nullptr, nullptr, Scross, nullptr,
                64, C_NOBJ, C_HD, C_EDIM, C_EDIM, C_NOBJ, sc, 0,
                C_NHEAD, C_NHEAD, 0, C_HD, 0, C_HD, 0, (long)64 * C_NOBJ);
    hipLaunchKernelGGL(softmax_rows, dim3(C_NHEAD * 64), dim3(256), 0, stream,
                       Scross, Pch, C_NOBJ, (const int*)nullptr, 1);

    // 4) V^T (4,256,4096); O64 = attn @ V_h, head-concat -> (64,1024) f16
    hipLaunchKernelGGL(transpose_v, dim3(C_HD * C_NOBJ / 256, C_NHEAD), dim3(256), 0,
                       stream, Vch, Vtc, C_EDIM, 0, 0, C_NOBJ, C_NHEAD);
    launch_gemm(stream, Pch, Vtc, nullptr, nullptr, nullptr, O64h,
                64, C_HD, C_NOBJ, C_NOBJ, C_NOBJ, C_EDIM, 1.0f, 0,
                C_NHEAD, C_NHEAD, 0, (long)64 * C_NOBJ, 0, (long)C_HD * C_NOBJ, 0, C_HD);

    // 5) cross out-proj -> per-frame fp32
    launch_gemm(stream, O64h, Wcout, cout_b, nullptr, X64, nullptr,
                64, C_EDIM, C_EDIM, C_EDIM, C_EDIM, C_EDIM, 1.0f, 0);

    // 6) obj_mod + window build (+pos emb, zero-pad invalid)
    hipLaunchKernelGGL(obj_mod_kernel, dim3(C_NOBJ), dim3(256), 0, stream,
                       obj, X64, obj_idx, OMod);
    hipLaunchKernelGGL(build_windows, dim3(C_ROWS), dim3(256), 0, stream,
                       OMod, pos_emb, iCum, Xbuf, Xh);

    // ---- transformer layers ----
    float*    xa  = Xbuf; _Float16* xah = Xh;
    float*    xb  = Ubuf; _Float16* xbh = Uh;
    const long S_z = (long)C_LWIN * C_LWIN;      // 36864
    const long V_z = (long)C_HD * C_LWIN;        // 49152
    for (int i = 0; i < 2; ++i) {
        const _Float16* inW  = Wsin  + (size_t)i * 3 * E * E;
        const float*    inB  = sa_in_b  + (size_t)i * 3 * E;
        const _Float16* outW = Wsout + (size_t)i * E * E;
        const float*    outB = sa_out_b + (size_t)i * E;
        const _Float16* l1W  = Wl1   + (size_t)i * C_DFF * E;
        const float*    l1B  = sa_l1_b  + (size_t)i * C_DFF;
        const _Float16* l2W  = Wl2   + (size_t)i * E * C_DFF;
        const float*    l2B  = sa_l2_b  + (size_t)i * E;

        // QKV projection: (11904, 3072) f16
        launch_gemm(stream, xah, inW, inB, nullptr, nullptr, QKVh,
                    C_ROWS, 3 * C_EDIM, C_EDIM, C_EDIM, C_EDIM, 3 * C_EDIM, 1.0f, 0);

        // scores per (window, head): (248,192,192) f32; softmax (masked) -> f16
        launch_gemm(stream, QKVh, QKVh + C_EDIM, nullptr, nullptr, Sbuf, nullptr,
                    C_LWIN, C_LWIN, C_HD, 3 * C_EDIM, 3 * C_EDIM, C_LWIN, sc, 0,
                    C_NW * C_NHEAD, C_NHEAD,
                    (long)C_LWIN * 3 * C_EDIM, C_HD,
                    (long)C_LWIN * 3 * C_EDIM, C_HD,
                    (long)C_NHEAD * S_z, S_z);
        hipLaunchKernelGGL(softmax_rows, dim3(C_NW * C_NHEAD * C_LWIN), dim3(256), 0,
                           stream, Sbuf, Pw, C_LWIN, iLen, C_NHEAD * C_LWIN);

        // V^T per (w,h): (248,256,192); attn@V -> head-concat (11904,1024) f16
        hipLaunchKernelGGL(transpose_v, dim3(C_HD * C_LWIN / 256, C_NW * C_NHEAD),
                           dim3(256), 0, stream,
                           QKVh, Vt, 3 * C_EDIM, C_LWIN, 2 * C_EDIM, C_LWIN, C_NHEAD);
        launch_gemm(stream, Pw, Vt, nullptr, nullptr, nullptr, Obh,
                    C_LWIN, C_HD, C_LWIN, C_LWIN, C_LWIN, C_EDIM, 1.0f, 0,
                    C_NW * C_NHEAD, C_NHEAD,
                    (long)C_NHEAD * S_z, S_z,
                    (long)C_NHEAD * V_z, V_z,
                    (long)C_LWIN * C_EDIM, C_HD);

        // out proj + residual -> f32; LayerNorm1 in-place (+f16 copy)
        launch_gemm(stream, Obh, outW, outB, xa, Tbuf, nullptr,
                    C_ROWS, C_EDIM, C_EDIM, C_EDIM, C_EDIM, C_EDIM, 1.0f, 0);
        hipLaunchKernelGGL(layernorm_rows, dim3(C_ROWS), dim3(256), 0, stream,
                           Tbuf, Th, sa_n1_g + (size_t)i * E, sa_n1_b + (size_t)i * E, C_EDIM);

        // FFN: H1 = relu(T @ l1^T + b1) f16; xb = H1 @ l2^T + b2 + T; LN2 (+f16)
        launch_gemm(stream, Th, l1W, l1B, nullptr, nullptr, H1h,
                    C_ROWS, C_DFF, C_EDIM, C_EDIM, C_EDIM, C_DFF, 1.0f, 1);
        launch_gemm(stream, H1h, l2W, l2B, Tbuf, xb, nullptr,
                    C_ROWS, C_EDIM, C_DFF, C_DFF, C_DFF, C_EDIM, 1.0f, 0);
        hipLaunchKernelGGL(layernorm_rows, dim3(C_ROWS), dim3(256), 0, stream,
                           xb, xbh, sa_n2_g + (size_t)i * E, sa_n2_b + (size_t)i * E, C_EDIM);

        float* tf = xa; xa = xb; xb = tf;
        _Float16* th = xah; xah = xbh; xbh = th;
    }

    // final gather: out[i] = X[win, slot]
    hipLaunchKernelGGL(gather_out, dim3(C_NOBJ), dim3(256), 0, stream,
                       xa, obj_idx, iCum, out);
}
